// NasRNN_75196287418745
// MI455X (gfx1250) — matrix operand order repaired and checked by reading, not compiled, not run on hardware
//
#include <hip/hip_runtime.h>

// ---------------------------------------------------------------------------
// NAS-RNN scan for gfx1250 (MI455X):
//  - fused input+recurrent projection, bf16 WMMA, register-resident weights
//  - TDM stages x_s / state_m tiles into padded LDS rows; x prefetched one
//    step ahead; state DMA overlapped with the X GEMM pass (split waits)
//  - software-pipelined A-fragment ds_loads (rotating registers)
//  - 16-WG cluster, per-step cluster barrier + atomic fallback
// ---------------------------------------------------------------------------

typedef __attribute__((ext_vector_type(16))) __bf16 v16bf;
typedef __attribute__((ext_vector_type(8)))  float  v8f;
typedef __attribute__((ext_vector_type(4)))  unsigned int v4u;
typedef __attribute__((ext_vector_type(8)))  int    v8i;
typedef __attribute__((ext_vector_type(4)))  int    v4i;

#define SEQ    1024
#define BATCH  64
#define IDIM   256
#define HDIM   256
#define NGATE  8
#define NSTRIP 16          // HDIM / 16 column strips
#define NKC    8           // 256 / 32 K-chunks
#define NBLK   16          // workgroups (== cluster size, one per WGP)

// LDS tile: 64 rows x 256 bf16, each 512B row padded by 16B -> stride 528B.
#define ROW_U4      33                     // uint4 per padded row
#define TILE_U4     (BATCH * ROW_U4)       // 2112 uint4 = 33792 B
#define TILE_BYTES  (TILE_U4 * 16)
#define GATE_F32    (NGATE * BATCH * 16)   // 8192 floats = 32 KB
#define SMEM_BYTES  (3 * TILE_BYTES + GATE_F32 * 4)   // 134144 B

#if defined(__has_builtin)
#if __has_builtin(__builtin_amdgcn_tensor_load_to_lds) && \
    __has_builtin(__builtin_amdgcn_s_wait_tensorcnt)
#define HAVE_TDM 1
#endif
#endif
#ifndef HAVE_TDM
#define HAVE_TDM 0
#endif

union FragU { uint4 u[2]; v16bf v; };

// LDS byte offset of a shared-memory pointer (low 32 bits of the flat
// address are the LDS byte address per the CDNA5 aperture mapping).
__device__ inline unsigned lds_off(const void* p) {
  return (unsigned)(uintptr_t)p;
}

// A-matrix fragment (16x32 bf16) from a padded LDS tile (stride ROW_U4).
// Lanes 0-15 / 16-31 hold M=0..15 with K-halves {0..7,16..23}/{8..15,24..31}.
__device__ __attribute__((always_inline)) inline
v16bf load_a_frag(const uint4* rowbase, int row, int kc, int lane) {
  const int half = (lane >> 4) & 1;
  const int i0 = row * ROW_U4 + kc * 4 + half;
  FragU f;
  f.u[0] = rowbase[i0];
  f.u[1] = rowbase[i0 + 2];   // +16 elements (K+16)
  return f.v;
}

// B-matrix fragment from pre-swizzled weights: [lane][16 bf16 contiguous].
__device__ inline v16bf load_b_frag(const uint4* __restrict__ swz,
                                    int fragIdx, int lane) {
  const uint4* p = swz + fragIdx * 64 + lane * 2;
  FragU f;
  f.u[0] = p[0];
  f.u[1] = p[1];
  return f.v;
}

// One GEMM pass: acc[t] += A(rows) x B(wreg) over all 8 K-chunks, with the
// next chunk's 4 A-fragments prefetched into rotating registers while the
// current chunk's 4 WMMAs issue (hides ds_load latency behind WMMA).
__device__ __attribute__((always_inline)) inline
void gemm_pass(const uint4* rows, const v16bf (&wreg)[NKC],
               v8f (&acc)[4], int lane) {
  const int r15 = lane & 15;
  v16bf a[4];
#pragma unroll
  for (int t = 0; t < 4; ++t) a[t] = load_a_frag(rows, t * 16 + r15, 0, lane);
#pragma unroll
  for (int kc = 0; kc < NKC; ++kc) {
    v16bf nx[4];
    if (kc + 1 < NKC) {
#pragma unroll
      for (int t = 0; t < 4; ++t)
        nx[t] = load_a_frag(rows, t * 16 + r15, kc + 1, lane);
    }
    const v16bf b = wreg[kc];
#pragma unroll
    for (int t = 0; t < 4; ++t)
      acc[t] = __builtin_amdgcn_wmma_f32_16x16x32_bf16(
          false, a[t], false, b, (short)0, acc[t], false, false);
    if (kc + 1 < NKC) {
#pragma unroll
      for (int t = 0; t < 4; ++t) a[t] = nx[t];
    }
  }
}

__device__ inline float sigm(float x) { return 1.0f / (1.0f + __expf(-x)); }

// Cross-workgroup barrier: CDNA5 cluster barrier (s_barrier_signal/-wait -3)
// plus device-scope atomic counter fallback for non-cluster dispatch.
__device__ inline void grid_barrier(unsigned* __restrict__ ctr, unsigned round) {
  __threadfence();
  __syncthreads();
  if (threadIdx.x == 0) {
    __hip_atomic_fetch_add(ctr, 1u, __ATOMIC_ACQ_REL, __HIP_MEMORY_SCOPE_AGENT);
    const unsigned goal = (round + 1u) * (unsigned)NBLK;
    while (__hip_atomic_load(ctr, __ATOMIC_ACQUIRE, __HIP_MEMORY_SCOPE_AGENT) < goal) {
      __builtin_amdgcn_s_sleep(1);
    }
  }
  __builtin_amdgcn_s_cluster_barrier();
  __syncthreads();
  __threadfence();
}

#if HAVE_TDM
// TDM: DMA a [64 x 256] bf16 row-major tile from global into LDS with 16 B
// padding after each 512 B row (pad_interval=128 DWORDs, pad_amount=4 DWORDs).
__device__ inline void tdm_load_tile(unsigned lds_byte, const void* gptr) {
  const unsigned long long ga = (unsigned long long)(uintptr_t)gptr;
  const v4u g0 = {
      1u,                                   // count=1, user descriptor
      lds_byte,                             // LDS byte address
      (unsigned)ga,                         // global_addr[31:0]
      ((unsigned)(ga >> 32) & 0x01FFFFFFu)  // global_addr[56:32]
          | 0x80000000u                     // type=2 ("image")
  };
  const v8i g1 = {
      (int)((1u << 16)      // data_size = 2 bytes
            | (1u << 20)    // pad_enable
            | (6u << 22)    // pad_interval: 128 DWORDs (one 512B row)
            | (3u << 25)),  // pad_amount: 4 DWORDs (16 B)
      (int)(256u << 16),    // tensor_dim0[15:0] = 256 (elements per row)
      (int)(64u << 16),     // tensor_dim0[31:16]=0 | tensor_dim1[15:0] = 64
      (int)(256u << 16),    // tensor_dim1[31:16]=0 | tile_dim0 = 256
      64,                   // tile_dim1 = 64, tile_dim2 = 0
      256,                  // tensor_dim0_stride = 256 elements
      0, 0
  };
  const v4i g2 = {0, 0, 0, 0};
  const v4i g3 = {0, 0, 0, 0};
#if defined(__clang_major__) && __clang_major__ >= 23
  const v8i gz = {0, 0, 0, 0, 0, 0, 0, 0};
  __builtin_amdgcn_tensor_load_to_lds(g0, g1, g2, g3, gz, 0);
#else
  __builtin_amdgcn_tensor_load_to_lds(g0, g1, g2, g3, 0);
#endif
}
#else
// Fallback: cooperative copy of a [64 x 256] bf16 tile into padded LDS rows.
__device__ inline void stage_tile(uint4* dst, const uint4* __restrict__ src,
                                  int tid) {
#pragma unroll
  for (int k = 0; k < 8; ++k) {
    const int idx = tid + 256 * k;       // 0..2047
    const int row = idx >> 5, col = idx & 31;
    dst[row * ROW_U4 + col] = src[row * 32 + col];
  }
}
#endif

// ---------------------------------------------------------------------------
// Prep kernels
// ---------------------------------------------------------------------------

__global__ void convert_x_bf16(const float4* __restrict__ x,
                               uint2* __restrict__ y, int n4) {
  const int i = blockIdx.x * blockDim.x + threadIdx.x;
  if (i >= n4) return;
  const float4 v = x[i];
  union { __bf16 h[4]; uint2 u; } o;
  o.h[0] = (__bf16)v.x; o.h[1] = (__bf16)v.y;
  o.h[2] = (__bf16)v.z; o.h[3] = (__bf16)v.w;
  y[i] = o.u;
}

// Swizzle weight tensor [8][K=256][N=256] f32 into B-fragment order:
// frag = ((g*16 + strip)*8 + kc), then [lane][16 contiguous bf16] where
// lane 0-15 = N col (K 0..7,16..23 of chunk), lane 16-31 = same cols (K 8..15,24..31).
__global__ void swizzle_w(const float* __restrict__ W, __bf16* __restrict__ dst) {
  const int o     = blockIdx.x * 256 + threadIdx.x;  // 0 .. 524287
  const int v     = o & 15;
  const int lane  = (o >> 4) & 31;
  const int frag  = o >> 9;
  const int kc    = frag & 7;
  const int gs    = frag >> 3;
  const int strip = gs & 15;
  const int g     = gs >> 4;
  const int k     = kc * 32 + ((v >> 3) << 4) + ((lane >> 4) << 3) + (v & 7);
  const int n     = strip * 16 + (lane & 15);
  dst[o] = (__bf16)W[(g * 256 + k) * 256 + n];
}

// ---------------------------------------------------------------------------
// The scan: 16 WGs x 256 threads; WG = one 16-wide H strip, wave = one gate.
// ---------------------------------------------------------------------------

__global__ void __launch_bounds__(256, 1)
nas_scan(const __bf16* __restrict__ Xb,    // [SEQ][64][256] bf16
         const uint4* __restrict__ swzIh,  // swizzled W_ih fragments
         const uint4* __restrict__ swzHh,  // swizzled W_hh fragments
         __bf16* __restrict__ st0,         // state_m ping  [64][256] bf16
         __bf16* __restrict__ st1,         // state_m pong
         float* __restrict__ out,          // [64][256] f32
         unsigned* __restrict__ ctr) {
  const int strip = blockIdx.x;   // 0..15
  const int tid   = threadIdx.x;  // 0..255
  const int gate  = tid >> 5;     // wave id == gate id
  const int lane  = tid & 31;

  // Single LDS base; all views are index offsets from it so clang's
  // infer-address-spaces keeps every access on the ds_ path (no flat_load).
  extern __shared__ __align__(16) unsigned char smem[];
  uint4* const lds_x0 = (uint4*)smem;                // x tiles: [2][TILE_U4]
  uint4* const lds_m  = lds_x0 + 2 * TILE_U4;        // state_m tile
  float* const lds_g  = (float*)(lds_x0 + 3 * TILE_U4);  // gate exchange

#if HAVE_TDM
  // Prefetch x_0 immediately (completion enforced by step-0 tensorcnt wait).
  if (gate == 0) tdm_load_tile(lds_off(lds_x0), Xb);
#endif

  // Zero our strip of the initial state.
#pragma unroll
  for (int j = 0; j < 4; ++j) {
    const int e = tid + 256 * j;
    const int m = e >> 4, n = e & 15;
    st0[m * HDIM + strip * 16 + n] = (__bf16)0.0f;
  }
  grid_barrier(ctr, 0u);

  // Register-resident weight fragments for this (gate, strip): 2 x 64 VGPRs.
  v16bf wih[NKC], whh[NKC];
#pragma unroll
  for (int kc = 0; kc < NKC; ++kc) {
    const int f = (gate * NSTRIP + strip) * NKC + kc;
    wih[kc] = load_b_frag(swzIh, f, lane);
    whh[kc] = load_b_frag(swzHh, f, lane);
  }

  float c[4]  = {0.f, 0.f, 0.f, 0.f};  // state_c (this thread's 4 elements)
  float mv[4] = {0.f, 0.f, 0.f, 0.f};  // last new_m

  for (int s = 0; s < SEQ; ++s) {
    const __bf16* stCur = (s & 1) ? st1 : st0;
    __bf16*       stNxt = (s & 1) ? st0 : st1;
    const uint4* xrows = lds_x0 + (s & 1) * TILE_U4;

    const v8f vz = {0.f, 0.f, 0.f, 0.f, 0.f, 0.f, 0.f, 0.f};
    v8f acc[4];
    acc[0] = vz; acc[1] = vz; acc[2] = vz; acc[3] = vz;

#if HAVE_TDM
    // Issue state DMA + x_{s+1} prefetch. TDM completes in order, so:
    //   tensorcnt<=2  => x_s resident      (start X pass; state DMA overlaps)
    //   tensorcnt<=1  => state_s resident  (start M pass; x_{s+1} in flight)
    if (gate == 0) {
      tdm_load_tile(lds_off(lds_m), stCur);
      if (s + 1 < SEQ) {
        tdm_load_tile(lds_off(lds_x0 + ((s + 1) & 1) * TILE_U4),
                      Xb + (size_t)(s + 1) * (BATCH * IDIM));
        __builtin_amdgcn_s_wait_tensorcnt(2);
      } else {
        __builtin_amdgcn_s_wait_tensorcnt(1);
      }
    }
    __syncthreads();
    gemm_pass(xrows, wih, acc, lane);        // overlaps state DMA
    if (gate == 0) {
      if (s + 1 < SEQ) __builtin_amdgcn_s_wait_tensorcnt(1);
      else             __builtin_amdgcn_s_wait_tensorcnt(0);
    }
    __syncthreads();
    gemm_pass(lds_m, whh, acc, lane);        // x_{s+1} DMA still in flight
#else
    stage_tile(lds_x0 + (s & 1) * TILE_U4,
               (const uint4*)(Xb + (size_t)s * (BATCH * IDIM)), tid);
    stage_tile(lds_m, (const uint4*)stCur, tid);
    __syncthreads();
    gemm_pass(xrows, wih, acc, lane);
    gemm_pass(lds_m, whh, acc, lane);
#endif

    // Spill gate block [64 x 16] to LDS for the cross-gate elementwise phase.
#pragma unroll
    for (int t = 0; t < 4; ++t) {
#pragma unroll
      for (int i = 0; i < 8; ++i) {
        const int m = t * 16 + i + ((lane >> 4) << 3);
        const int n = lane & 15;
        lds_g[(gate * 64 + m) * 16 + n] = acc[t][i];
      }
    }
    __syncthreads();

    // NAS-cell nonlinearity chain; 4 (b,h) elements per thread.
#pragma unroll
    for (int j = 0; j < 4; ++j) {
      const int e = tid + 256 * j;  // e = m*16 + n within the strip
      const float g0 = lds_g[0 * 1024 + e];
      const float g1 = lds_g[1 * 1024 + e];
      const float g2 = lds_g[2 * 1024 + e];
      const float g3 = lds_g[3 * 1024 + e];
      const float g4 = lds_g[4 * 1024 + e];
      const float g5 = lds_g[5 * 1024 + e];
      const float g6 = lds_g[6 * 1024 + e];
      const float g7 = lds_g[7 * 1024 + e];
      const float l1_0 = sigm(g0);
      const float l1_1 = fmaxf(g1, 0.0f);
      const float l1_2 = sigm(g2);
      const float l1_3 = fmaxf(g3, 0.0f);
      const float l1_4 = tanhf(g4);
      const float l1_5 = sigm(g5);
      const float l1_6 = tanhf(g6);
      const float l1_7 = sigm(g7);
      const float l2_0 = tanhf(l1_0 * l1_1);
      const float l2_1 = tanhf(l1_2 + l1_3);
      const float l2_2 = tanhf(l1_4 * l1_5);
      const float l2_3 = sigm(l1_6 + l1_7);
      const float l2_0v2 = tanhf(l2_0 + c[j]);
      const float nc = l2_0v2 * l2_1;
      const float l3_1 = tanhf(l2_2 + l2_3);
      const float nm = tanhf(nc * l3_1);
      c[j]  = nc;
      mv[j] = nm;
      const int m = e >> 4, n = e & 15;
      stNxt[m * HDIM + strip * 16 + n] = (__bf16)nm;
    }
    // Publish new_m strip, sync entire cluster before next step reads it.
    grid_barrier(ctr, (unsigned)(s + 1));
  }

  // out = state_c + state_m
#pragma unroll
  for (int j = 0; j < 4; ++j) {
    const int e = tid + 256 * j;
    const int m = e >> 4, n = e & 15;
    out[m * HDIM + strip * 16 + n] = c[j] + mv[j];
  }
}

// ---------------------------------------------------------------------------
// Launch
// ---------------------------------------------------------------------------

extern "C" void kernel_launch(void* const* d_in, const int* in_sizes, int n_in,
                              void* d_out, int out_size, void* d_ws, size_t ws_size,
                              hipStream_t stream) {
  (void)in_sizes; (void)n_in; (void)out_size; (void)ws_size;

  const float* x   = (const float*)d_in[0];  // [1024, 64, 256]
  const float* wih = (const float*)d_in[1];  // [8, 256, 256]
  const float* whh = (const float*)d_in[2];  // [8, 256, 256]
  float* out = (float*)d_out;                // [64, 256]

  char* ws = (char*)d_ws;
  size_t off = 0;
  __bf16* Xb  = (__bf16*)(ws + off); off += (size_t)SEQ * BATCH * IDIM * 2;   // 32 MB
  __bf16* sIh = (__bf16*)(ws + off); off += (size_t)NGATE * IDIM * HDIM * 2;  // 1 MB
  __bf16* sHh = (__bf16*)(ws + off); off += (size_t)NGATE * HDIM * HDIM * 2;  // 1 MB
  __bf16* st0 = (__bf16*)(ws + off); off += (size_t)BATCH * HDIM * 2;         // 32 KB
  __bf16* st1 = (__bf16*)(ws + off); off += (size_t)BATCH * HDIM * 2;         // 32 KB
  unsigned* ctr = (unsigned*)(ws + off); off += 256;

  // Barrier counter must start at 0 every call (graph-capture safe).
  hipMemsetAsync(ctr, 0, sizeof(unsigned), stream);

  const int n4 = SEQ * BATCH * IDIM / 4;
  convert_x_bf16<<<(n4 + 255) / 256, 256, 0, stream>>>((const float4*)x,
                                                       (uint2*)Xb, n4);
  swizzle_w<<<(NGATE * IDIM * HDIM) / 256, 256, 0, stream>>>(wih, sIh);
  swizzle_w<<<(NGATE * HDIM * HDIM) / 256, 256, 0, stream>>>(whh, sHh);

  // >64 KB dynamic LDS: raise the limit (no-op where unnecessary).
  hipFuncSetAttribute((const void*)nas_scan,
                      hipFuncAttributeMaxDynamicSharedMemorySize, SMEM_BYTES);

  nas_scan<<<NBLK, 256, SMEM_BYTES, stream>>>(Xb, (const uint4*)sIh,
                                              (const uint4*)sHh, st0, st1,
                                              out, ctr);
}
